// PrototypeResidualAdapter_46720654246146
// MI455X (gfx1250) — matrix operand order repaired and checked by reading, not compiled, not run on hardware
//
#include <hip/hip_runtime.h>
#include <hip/hip_bf16.h>
#include <math.h>

// ---------------------------------------------------------------------------
// Cluster-routed residual adapters on gfx1250 (MI455X).
//   out[b] = h[b] + W2[k]·gelu(W1[k]·h[b] + b1[k]) + b2[k],  k = ids[b]
// B=32768, D=1024, C=128, K=16.
// Strategy: bucket rows by cluster, then grouped bf16 WMMA GEMM per
// (cluster, 16-row tile): LDS-staged bf16 h tile -> v_wmma_f32_16x16x32_bf16
// down-proj -> gelu -> LDS bf16 mid -> wmma up-proj -> residual store.
// ---------------------------------------------------------------------------

#define BB 32768
#define DD 1024
#define CC 128
#define KK 16

#define HSTR 1032   // bf16 elems per LDS h row (pad: 2064B = 516 dw, rotates banks)
#define MSTR 136    // bf16 elems per LDS mid row (272B = 68 dw)

typedef __attribute__((ext_vector_type(16))) __bf16 v16bf;
typedef __attribute__((ext_vector_type(8)))  __bf16 v8bf;
typedef __attribute__((ext_vector_type(8)))  float  v8f;

// A-fragment (16x32 bf16): per lane, 16 contiguous K elements (lanes 0-15
// hold K=0..15, lanes 16-31 hold K=16..31), read as two 8-element chunks.
static __device__ __forceinline__ v16bf load_a_frag(const __bf16* p) {
    v8bf lo = *(const v8bf*)(p);
    v8bf hi = *(const v8bf*)(p + 8);
    return __builtin_shufflevector(lo, hi, 0, 1, 2, 3, 4, 5, 6, 7,
                                           8, 9, 10, 11, 12, 13, 14, 15);
}

// B-fragment (32x16 bf16): per lane, 16 contiguous K elements (lanes 0-15
// hold K=0..15, lanes 16-31 hold K=16..31) -> single 32B contiguous read.
static __device__ __forceinline__ v16bf load_b_frag(const __bf16* p) {
    return *(const v16bf*)(p);
}

static __device__ __forceinline__ float gelu_erf(float x) {
    return 0.5f * x * (1.0f + erff(x * 0.70710678118654752440f));
}

// ---------------------------------------------------------------------------
// Prep: convert W1/W2 fp32 -> bf16 once per launch (weights then live in L2).
// ---------------------------------------------------------------------------
__global__ void __launch_bounds__(256)
convert_w_kernel(const float* __restrict__ w1, const float* __restrict__ w2,
                 __bf16* __restrict__ w1b, __bf16* __restrict__ w2b, int n) {
    int i = blockIdx.x * blockDim.x + threadIdx.x;
    int stride = gridDim.x * blockDim.x;
    for (; i < 2 * n; i += stride) {
        if (i < n) w1b[i] = (__bf16)w1[i];
        else       w2b[i - n] = (__bf16)w2[i - n];
    }
}

__global__ void zero_cnt_kernel(int* __restrict__ cnt) {
    if (threadIdx.x < KK) cnt[threadIdx.x] = 0;
}

__global__ void __launch_bounds__(256)
gather_kernel(const int* __restrict__ ids, int* __restrict__ cnt,
              int* __restrict__ rowlist) {
    int b = blockIdx.x * blockDim.x + threadIdx.x;
    if (b < BB) {
        int k = ids[b];
        int pos = atomicAdd(&cnt[k], 1);
        rowlist[k * BB + pos] = b;
    }
}

// ---------------------------------------------------------------------------
// Main grouped-GEMM adapter kernel. Block = 256 threads = 8 waves (wave32).
// blockIdx.x = cluster k, blockIdx.y = 16-row tile within cluster's row list.
// ---------------------------------------------------------------------------
__global__ void __launch_bounds__(256)
adapter_kernel(const float* __restrict__ h,
               const float* __restrict__ b1,
               const float* __restrict__ b2,
               const __bf16* __restrict__ w1b,
               const __bf16* __restrict__ w2b,
               const int* __restrict__ cnt,
               const int* __restrict__ rowlist,
               float* __restrict__ out) {
    __shared__ __bf16 lds_h[16 * HSTR];     // 33 KB: 16 rows of h, bf16
    __shared__ __bf16 lds_mid[16 * MSTR];   // 4.3 KB: gelu(mid), bf16
    __shared__ int    lrows[16];

    const int k     = blockIdx.x;
    const int nrows = cnt[k];
    const int r0    = blockIdx.y * 16;
    if (r0 >= nrows) return;

    const int tid = threadIdx.x;

    if (tid < 16) {
        int pos = r0 + tid;
        lrows[tid] = rowlist[k * BB + (pos < nrows ? pos : r0)];
    }
    __syncthreads();

    // ---- Stage 16 gathered rows of h into LDS as bf16 (coalesced float4) ----
    for (int idx = tid; idx < 16 * (DD / 4); idx += 256) {
        int r   = idx >> 8;          // 256 float4 per row
        int col = (idx & 255) * 4;
        const float4 v = *(const float4*)(h + (size_t)lrows[r] * DD + col);
        __bf16* dst = &lds_h[r * HSTR + col];
        dst[0] = (__bf16)v.x; dst[1] = (__bf16)v.y;
        dst[2] = (__bf16)v.z; dst[3] = (__bf16)v.w;
    }
    __syncthreads();

    const int lane = tid & 31;
    const int wv   = tid >> 5;       // wave id 0..7
    const int half = lane >> 4;      // K-group select
    const int ln   = lane & 15;      // row (A) / col (B,C,D) within tile

    const __bf16* w1k = w1b + (size_t)k * CC * DD;  // [C][D], contiguous in d
    const __bf16* w2k = w2b + (size_t)k * DD * CC;  // [D][C], contiguous in c

    // ---- Down-proj: wave wv owns mid cols [wv*16, wv*16+16). K-dim = D. ----
    {
        const int n0 = wv * 16;
        v8f acc = {};
        const __bf16* bbase = w1k + (size_t)(n0 + ln) * DD + half * 16;
        const __bf16* abase = &lds_h[ln * HSTR + half * 16];
#pragma unroll 4
        for (int kk = 0; kk < DD; kk += 32) {
            v16bf a = load_a_frag(abase + kk);
            v16bf bm = load_b_frag(bbase + kk);
            acc = __builtin_amdgcn_wmma_f32_16x16x32_bf16(
                false, a, false, bm, (short)0, acc, false, false);
        }
        const float bias = b1[k * CC + n0 + ln];
#pragma unroll
        for (int j = 0; j < 8; ++j) {
            int m = j + half * 8;                    // C/D layout: M = j + 8*half
            float x = gelu_erf(acc[j] + bias);
            lds_mid[m * MSTR + n0 + ln] = (__bf16)x;
        }
    }
    __syncthreads();

    // ---- Up-proj: wave wv owns 8 output D-tiles. K-dim = C (4 WMMA steps). ----
    v16bf amid[4];
#pragma unroll
    for (int c = 0; c < 4; ++c)
        amid[c] = load_a_frag(&lds_mid[ln * MSTR + c * 32 + half * 16]);

    for (int i = 0; i < 8; ++i) {
        const int n0 = (wv * 8 + i) * 16;
        v8f acc = {};
        const __bf16* bbase = w2k + (size_t)(n0 + ln) * CC + half * 16;
#pragma unroll
        for (int c = 0; c < 4; ++c) {
            v16bf bm = load_b_frag(bbase + c * 32);
            acc = __builtin_amdgcn_wmma_f32_16x16x32_bf16(
                false, amid[c], false, bm, (short)0, acc, false, false);
        }
        const float bias2 = b2[k * DD + n0 + ln];
#pragma unroll
        for (int j = 0; j < 8; ++j) {
            int m = j + half * 8;
            if (r0 + m < nrows) {                    // mask padded tail rows
                size_t off = (size_t)lrows[m] * DD + n0 + ln;
                out[off] = h[off] + acc[j] + bias2;  // residual in fp32
            }
        }
    }
}

// ---------------------------------------------------------------------------
// Launch. Inputs: h, cluster_ids, W1, b1, W2, b2. Output: (B, D) fp32.
// Workspace: [0,4MB) W1 bf16 | [4MB,8MB) W2 bf16 | cnt | rowlist (2MB).
// ---------------------------------------------------------------------------
extern "C" void kernel_launch(void* const* d_in, const int* in_sizes, int n_in,
                              void* d_out, int out_size, void* d_ws, size_t ws_size,
                              hipStream_t stream) {
    const float* h   = (const float*)d_in[0];
    const int*   ids = (const int*)d_in[1];
    const float* W1  = (const float*)d_in[2];
    const float* b1  = (const float*)d_in[3];
    const float* W2  = (const float*)d_in[4];
    const float* b2  = (const float*)d_in[5];
    float*       out = (float*)d_out;

    const int    NW     = KK * CC * DD;              // 2,097,152 elems per W
    const size_t WBYTES = (size_t)NW * sizeof(__bf16);

    char*   ws      = (char*)d_ws;
    __bf16* w1b     = (__bf16*)(ws);
    __bf16* w2b     = (__bf16*)(ws + WBYTES);
    int*    cnt     = (int*)(ws + 2 * WBYTES);
    int*    rowlist = (int*)(ws + 2 * WBYTES + 256);

    convert_w_kernel<<<2048, 256, 0, stream>>>(W1, W2, w1b, w2b, NW);
    zero_cnt_kernel<<<1, 32, 0, stream>>>(cnt);
    gather_kernel<<<BB / 256, 256, 0, stream>>>(ids, cnt, rowlist);

    dim3 grid(KK, BB / 16);   // over-launched; blocks past a cluster's tile count exit
    adapter_kernel<<<grid, 256, 0, stream>>>(h, b1, b2, w1b, w2b, cnt, rowlist, out);
}
